// GRUCellModel_54039278518362
// MI455X (gfx1250) — matrix operand order repaired and checked by reading, not compiled, not run on hardware
//
#include <hip/hip_runtime.h>

// ---------------------------------------------------------------------------
// GRU classifier for MI455X (gfx1250, wave32, WMMA).
//   B=512, T=256, E=H=512, 3H=1536, VOCAB=30, CLASSES=2
//
// Strategy:
//   prep1: gi_table[v][j] = embed[v,:] . w_ih[j,:] + b_ih[j]   (30 x 1536 f32)
//   prep2: whh_b = bf16(w_hh)                                  (1536 x 512)
//   main : 32 WGs x 512 thr; WG owns 16 batch rows; per step
//          gh = h(16x512,bf16) x whh^T via v_wmma_f32_16x16x32_bf16,
//          gates fused with gi_table gather + valid mask, h kept in LDS.
// ---------------------------------------------------------------------------

#define VOCAB   30
#define EMBED   512
#define HIDDEN  512
#define GATES   1536
#define BATCH   512
#define SEQ     256
#define ROWS    16      // batch rows per workgroup (WMMA M)
#define THREADS 512     // 16 waves of 32
#define WAVES   16
#define NT      6       // N-tiles per wave: 96 tiles / 16 waves
// LDS row strides padded to 1040/2064 bytes (odd multiple of 16B) to rotate banks
#define HF_STRIDE 516   // floats
#define HB_STRIDE 520   // bf16 (ushort)
#define X_STRIDE  260   // ints

typedef __attribute__((ext_vector_type(16))) __bf16       v16bf;
typedef __attribute__((ext_vector_type(8)))  float        v8f;
typedef __attribute__((ext_vector_type(4)))  unsigned int u32x4;

union FragBF { u32x4 q[2]; v16bf m; };

__device__ __forceinline__ unsigned short f32_to_bf16(float f) {
  unsigned u = __float_as_uint(f);
  u += 0x7FFFu + ((u >> 16) & 1u);          // round-to-nearest-even
  return (unsigned short)(u >> 16);
}
__device__ __forceinline__ float fast_sigmoid(float x) {
  return 1.0f / (1.0f + __expf(-x));
}
__device__ __forceinline__ float fast_tanh(float x) {
  return 1.0f - 2.0f / (__expf(2.0f * x) + 1.0f);  // saturates cleanly at +-1
}

// ---- prep 1: gi_table = embed @ w_ih^T + b_ih  (30 x 1536) ------------------
__global__ void gru_prep_gi(const float* __restrict__ embed,
                            const float* __restrict__ w_ih,
                            const float* __restrict__ b_ih,
                            float* __restrict__ gi_tab) {
  int idx = blockIdx.x * blockDim.x + threadIdx.x;
  if (idx >= VOCAB * GATES) return;
  int v = idx / GATES;
  int j = idx - v * GATES;
  const float4* e = (const float4*)(embed + v * EMBED);
  const float4* w = (const float4*)(w_ih + j * EMBED);
  float acc = b_ih[j];
#pragma unroll 4
  for (int k = 0; k < EMBED / 4; ++k) {
    float4 a = e[k], b = w[k];
    acc += a.x * b.x + a.y * b.y + a.z * b.z + a.w * b.w;
  }
  gi_tab[idx] = acc;
}

// ---- prep 2: w_hh -> bf16 ---------------------------------------------------
__global__ void gru_prep_whh(const float* __restrict__ w_hh,
                             unsigned short* __restrict__ whh_b) {
  int idx = blockIdx.x * blockDim.x + threadIdx.x;
  if (idx < GATES * HIDDEN) whh_b[idx] = f32_to_bf16(w_hh[idx]);
}

// ---- main recurrent kernel --------------------------------------------------
__global__ __launch_bounds__(THREADS, 1)
void gru_main(const int* __restrict__ x,
              const float* __restrict__ embed,
              const unsigned short* __restrict__ whh_b,
              const float* __restrict__ gi_tab,
              const float* __restrict__ b_hh,
              const float* __restrict__ fc_w,
              const float* __restrict__ fc_b,
              float* __restrict__ out) {
  __shared__ __align__(16) float          s_hf[ROWS * HF_STRIDE]; // h, f32
  __shared__ __align__(16) unsigned short s_hb[ROWS * HB_STRIDE]; // h, bf16
  __shared__ __align__(16) int            s_x [ROWS * X_STRIDE];  // token ids

  const int tid   = threadIdx.x;
  const int wave  = tid >> 5;
  const int lane  = tid & 31;
  const int lrow  = lane & 15;   // M row (A/C) or N col (B/C) within a tile
  const int khalf = lane >> 4;   // lane-half selects K sub-pattern
  const int b0    = blockIdx.x * ROWS;

  // stage this WG's token ids
  for (int i = tid; i < ROWS * SEQ; i += THREADS) {
    int r = i >> 8, t = i & (SEQ - 1);
    s_x[r * X_STRIDE + t] = x[(b0 + r) * SEQ + t];
  }
  // h0 = embed_table[x[:,0]]
  for (int i = tid; i < ROWS * HIDDEN; i += THREADS) {
    int r = i >> 9, k = i & (HIDDEN - 1);
    int v = x[(b0 + r) * SEQ];
    float hv = embed[v * EMBED + k];
    s_hf[r * HF_STRIDE + k] = hv;
    s_hb[r * HB_STRIDE + k] = f32_to_bf16(hv);
  }
  __syncthreads();

  const v8f vzero = {0.f, 0.f, 0.f, 0.f, 0.f, 0.f, 0.f, 0.f};

  for (int t = 1; t < SEQ; ++t) {
    v8f acc[NT];
#pragma unroll
    for (int n = 0; n < NT; ++n) acc[n] = vzero;

    // gh = h x w_hh^T : M=16, N-tiles wave+16*nt, K=512 in 16 steps of 32
#pragma unroll 2
    for (int kt = 0; kt < HIDDEN / 32; ++kt) {
      const int k0 = kt * 32;
      // A fragment (16-bit A 16x32 layout, ISA 7.12.2):
      //   lanes 0-15: K = k0+[0..7], k0+[16..23]; lanes 16-31: +8 / +24
      FragBF a;
      const unsigned short* ap = &s_hb[lrow * HB_STRIDE + k0 + (khalf << 3)];
      a.q[0] = *(const u32x4*)(ap);
      a.q[1] = *(const u32x4*)(ap + 16);
#pragma unroll
      for (int nt = 0; nt < NT; ++nt) {
        const int ncol = (wave + nt * WAVES) * 16 + lrow;  // w_hh row / B col
        // B fragment: lane holds column ncol, K = k0+8*khalf+[0..7], k0+16+8*khalf+[0..7]
        FragBF b;
        const unsigned short* bp = &whh_b[ncol * HIDDEN + k0 + (khalf << 3)];
        b.q[0] = *(const u32x4*)(bp);
        b.q[1] = *(const u32x4*)(bp + 16);
        acc[nt] = __builtin_amdgcn_wmma_f32_16x16x32_bf16(
            false, a.m, false, b.m, (short)0, acc[nt], false, false);
      }
    }
    __syncthreads();  // all waves done reading h before it is overwritten

    // Gates. Wave w owns hidden tiles w, w+16 -> (r,z,n) = acc[i], acc[i+2], acc[i+4]
#pragma unroll
    for (int i = 0; i < 2; ++i) {
      const int hc = (wave + i * WAVES) * 16 + lrow;  // hidden column 0..511
      const float bR = b_hh[hc];
      const float bZ = b_hh[hc + HIDDEN];
      const float bN = b_hh[hc + 2 * HIDDEN];
#pragma unroll
      for (int v = 0; v < 8; ++v) {
        const int row = v + (khalf << 3);             // C/D layout: M = v + 8*(lane/16)
        const int xv  = s_x[row * X_STRIDE + t];
        const float* gp = gi_tab + xv * GATES + hc;
        const float giR = gp[0];
        const float giZ = gp[HIDDEN];
        const float giN = gp[2 * HIDDEN];
        const float hp  = s_hf[row * HF_STRIDE + hc];
        const float rg = fast_sigmoid(acc[i][v]     + giR + bR);
        const float zg = fast_sigmoid(acc[i + 2][v] + giZ + bZ);
        const float ng = fast_tanh(giN + rg * (acc[i + 4][v] + bN));
        float hn = (1.0f - zg) * ng + zg * hp;
        hn = (xv == 0) ? hp : hn;                     // padding: freeze h
        s_hf[row * HF_STRIDE + hc] = hn;
        s_hb[row * HB_STRIDE + hc] = f32_to_bf16(hn);
      }
    }
    __syncthreads();
  }

  // logits = h_final @ fc_w^T + fc_b   (16 rows x 2 classes per WG)
  if (tid < ROWS * 2) {
    const int r = tid >> 1, c = tid & 1;
    float accv = fc_b[c];
    const float* w = fc_w + c * HIDDEN;
#pragma unroll 4
    for (int k = 0; k < HIDDEN; ++k) accv += s_hf[r * HF_STRIDE + k] * w[k];
    out[(b0 + r) * 2 + c] = accv;
  }
}

// ---------------------------------------------------------------------------
extern "C" void kernel_launch(void* const* d_in, const int* in_sizes, int n_in,
                              void* d_out, int out_size, void* d_ws, size_t ws_size,
                              hipStream_t stream) {
  (void)in_sizes; (void)n_in; (void)out_size; (void)ws_size;
  const int*   x    = (const int*)d_in[0];
  const float* emb  = (const float*)d_in[1];
  const float* w_ih = (const float*)d_in[2];
  const float* w_hh = (const float*)d_in[3];
  const float* b_ih = (const float*)d_in[4];
  const float* b_hh = (const float*)d_in[5];
  const float* fc_w = (const float*)d_in[6];
  const float* fc_b = (const float*)d_in[7];
  float* out = (float*)d_out;

  // workspace: [ whh_bf16 : 1536*512*2 B ][ gi_table : 30*1536*4 B ]
  unsigned short* whh_b = (unsigned short*)d_ws;
  float* gi_tab = (float*)((char*)d_ws + (size_t)(GATES * HIDDEN * 2));

  gru_prep_gi<<<(VOCAB * GATES + 255) / 256, 256, 0, stream>>>(emb, w_ih, b_ih, gi_tab);
  gru_prep_whh<<<(GATES * HIDDEN + 255) / 256, 256, 0, stream>>>(w_hh, whh_b);
  gru_main<<<BATCH / ROWS, THREADS, 0, stream>>>(x, emb, whh_b, gi_tab, b_hh,
                                                 fc_w, fc_b, out);
}
